// MHA_6004364279961
// MI455X (gfx1250) — matrix operand order repaired and checked by reading, not compiled, not run on hardware
//
#include <hip/hip_runtime.h>
#include <math.h>
#include <stdint.h>

#define DEVFN __device__ __forceinline__

typedef __attribute__((ext_vector_type(16))) __bf16        v16bf;
typedef __attribute__((ext_vector_type(8)))  float         v8f;
typedef __attribute__((ext_vector_type(4)))  unsigned int  u32x4;

enum { NB = 4, NM = 2048, ND = 1024, NH = 16, NK = 64, NV = 64 };

union FragBF {
    u32x4 q[2];
    v16bf v;
};
union Pack8 {
    u32x4 q;
    __bf16 us[8];
};

DEVFN v8f v8f_zero() {
    v8f z = {0.f, 0.f, 0.f, 0.f, 0.f, 0.f, 0.f, 0.f};
    return z;
}

// Two contiguous 16B chunks -> one 16x(bf16) WMMA operand fragment.
DEVFN v16bf load_frag16(const __bf16* p0, const __bf16* p1) {
    FragBF f;
    f.q[0] = *reinterpret_cast<const u32x4*>(p0);
    f.q[1] = *reinterpret_cast<const u32x4*>(p1);
    return f.v;
}

DEVFN v8f wmma_bf16(v16bf a, v16bf b, v8f c) {
    return __builtin_amdgcn_wmma_f32_16x16x32_bf16(
        /*neg_a=*/false, a, /*neg_b=*/false, b,
        /*c_mod=*/(short)0, c, /*reuse_a=*/false, /*reuse_b=*/false);
}

// ---------------------------------------------------------------------------
// Pack kernels: fp32 -> bf16 (+ transposes for contiguous B-operand loads).
// All index math is shifts/masks (kd = 64 compile-time).
// ---------------------------------------------------------------------------
__global__ void pack_x_kernel(const float* __restrict__ src,
                              __bf16* __restrict__ dst, int n) {
    for (int i = blockIdx.x * blockDim.x + threadIdx.x; i < n;
         i += gridDim.x * blockDim.x)
        dst[i] = (__bf16)src[i];
}

// src [H][D][64]  ->  dst [H][64][D]   (transposed weight for B operand)
__global__ void pack_w_kernel(const float* __restrict__ src,
                              __bf16* __restrict__ dst) {
    const int n = NH * ND * NK;  // 1M, all powers of two
    for (int i = blockIdx.x * blockDim.x + threadIdx.x; i < n;
         i += gridDim.x * blockDim.x) {
        const int h = i >> 16;          // / (ND*NK)
        const int d = (i >> 6) & 1023;  // / NK % ND
        const int k = i & 63;           // % NK
        dst[((size_t)(h << 6 | k) << 10) + d] = (__bf16)src[i];
    }
}

// src [H][D][V] -> dst [D][H*V]   (W^T for the output projection B operand)
__global__ void pack_wo_kernel(const float* __restrict__ src,
                               __bf16* __restrict__ dst) {
    const int n = NH * ND * NV;
    for (int i = blockIdx.x * blockDim.x + threadIdx.x; i < n;
         i += gridDim.x * blockDim.x) {
        const int h = i >> 16;          // / (ND*NV)
        const int d = (i >> 6) & 1023;  // / NV % ND
        const int v = i & 63;           // % NV
        dst[((size_t)d << 10) + (h << 6) + v] = (__bf16)src[i];
    }
}

// ---------------------------------------------------------------------------
// Fused QKV projection: per (b,h), [M x D] x [D x 64] -> Q,K row-major, V^T
// One wave owns a 16-row strip; 12 WMMAs per A-tile load.
// ---------------------------------------------------------------------------
__global__ __launch_bounds__(256) void qkv_kernel(
    const __bf16* __restrict__ xb,
    const __bf16* __restrict__ pqt,
    const __bf16* __restrict__ pkt,
    const __bf16* __restrict__ pvt,
    __bf16* __restrict__ qb,
    __bf16* __restrict__ kb,
    __bf16* __restrict__ vtb) {
    const int lane = threadIdx.x & 31;
    const int wave = threadIdx.x >> 5;
    const int ln   = lane & 15;
    const int half = lane >> 4;
    const int bh = blockIdx.y;
    const int b  = bh >> 4;
    const int h  = bh & 15;
    const int m0 = blockIdx.x * 128 + wave * 16;

    const __bf16* xrow = xb + (size_t)b * NM * ND + (size_t)(m0 + ln) * ND;
    const __bf16* wq = pqt + (size_t)h * NK * ND;
    const __bf16* wk = pkt + (size_t)h * NK * ND;
    const __bf16* wv = pvt + (size_t)h * NK * ND;

    v8f aq[4], ak[4], av[4];
#pragma unroll
    for (int t = 0; t < 4; ++t) { aq[t] = v8f_zero(); ak[t] = v8f_zero(); av[t] = v8f_zero(); }

    for (int d0 = 0; d0 < ND; d0 += 32) {
        v16bf a = load_frag16(xrow + d0 + half * 16, xrow + d0 + half * 16 + 8);
#pragma unroll
        for (int t = 0; t < 4; ++t) {
            const size_t roff = (size_t)(t * 16 + ln) * ND + d0 + half * 16;
            aq[t] = wmma_bf16(a, load_frag16(wq + roff, wq + roff + 8), aq[t]);
            ak[t] = wmma_bf16(a, load_frag16(wk + roff, wk + roff + 8), ak[t]);
            av[t] = wmma_bf16(a, load_frag16(wv + roff, wv + roff + 8), av[t]);
        }
    }

    const size_t qkbase = (size_t)(b * NH + h) * NM * NK;
    __bf16* qo = qb + qkbase;
    __bf16* ko = kb + qkbase;
#pragma unroll
    for (int t = 0; t < 4; ++t) {
        const int n = t * 16 + ln;
#pragma unroll
        for (int j = 0; j < 8; ++j) {
            const int m = m0 + j + 8 * half;
            qo[(size_t)m * NK + n] = (__bf16)aq[t][j];
            ko[(size_t)m * NK + n] = (__bf16)ak[t][j];
        }
        Pack8 p8;
#pragma unroll
        for (int j = 0; j < 8; ++j) p8.us[j] = (__bf16)av[t][j];
        __bf16* vo = vtb + ((size_t)(b * NH + h) * NV + n) * NM + m0 + 8 * half;
        *reinterpret_cast<u32x4*>(vo) = p8.q;  // contiguous transposed store
    }
}

// ---------------------------------------------------------------------------
// Causal flash attention: each wave owns 16 query rows, streams kv in 32-col
// blocks with online softmax. P bounced through per-wave LDS tile for the
// C-layout -> A-layout rearrangement before the P*V WMMAs.
// ---------------------------------------------------------------------------
__global__ __launch_bounds__(256) void attn_kernel(
    const __bf16* __restrict__ qb,
    const __bf16* __restrict__ kb,
    const __bf16* __restrict__ vtb,
    __bf16* __restrict__ ob) {
    __shared__ __bf16 lds_p[8 * 16 * 32];  // 1KB per wave

    const int lane = threadIdx.x & 31;
    const int wave = threadIdx.x >> 5;
    const int ln   = lane & 15;
    const int half = lane >> 4;
    const int bh = blockIdx.y;
    const int b  = bh >> 4;
    const int h  = bh & 15;
    const int m0 = blockIdx.x * 128 + wave * 16;

    const __bf16* qh  = qb + (size_t)(b * NH + h) * NM * NK;
    const __bf16* khp = kb + (size_t)(b * NH + h) * NM * NK;
    const __bf16* vh  = vtb + (size_t)(b * NH + h) * NV * NM;
    __bf16* pst = lds_p + wave * (16 * 32);

    // Q fragments (kept resident for the whole kv loop)
    const __bf16* qrow = qh + (size_t)(m0 + ln) * NK;
    const v16bf aq0 = load_frag16(qrow + half * 16, qrow + half * 16 + 8);
    const v16bf aq1 = load_frag16(qrow + 32 + half * 16, qrow + 32 + half * 16 + 8);

    v8f accO[4];
#pragma unroll
    for (int t = 0; t < 4; ++t) accO[t] = v8f_zero();
    float mst[8], lst[8];
#pragma unroll
    for (int j = 0; j < 8; ++j) { mst[j] = -INFINITY; lst[j] = 0.0f; }

    const int n_end = m0 + 16;  // causal: only kv columns <= last row
    for (int n0 = 0; n0 < n_end; n0 += 32) {
        if (n0 + 32 < NM)
            __builtin_prefetch(khp + (size_t)(n0 + 32) * NK, 0, 1);

        // S = Q * K^T  (two 16x16 n-tiles, K-depth 64 = 2 WMMAs each)
        v8f s[2];
#pragma unroll
        for (int t = 0; t < 2; ++t) {
            const __bf16* krow = khp + (size_t)(n0 + t * 16 + ln) * NK;
            v8f st = v8f_zero();
            st = wmma_bf16(aq0, load_frag16(krow + half * 16, krow + half * 16 + 8), st);
            st = wmma_bf16(aq1, load_frag16(krow + 32 + half * 16, krow + 32 + half * 16 + 8), st);
            s[t] = st;
        }

        // scale + causal mask (lane = column, VGPR j = row j+8*half)
#pragma unroll
        for (int t = 0; t < 2; ++t) {
            const int ncol = n0 + t * 16 + ln;
#pragma unroll
            for (int j = 0; j < 8; ++j) {
                const float v = s[t][j] * 0.125f;  // 1/sqrt(64)
                const int mrow = m0 + j + 8 * half;
                s[t][j] = (ncol > mrow) ? -INFINITY : v;
            }
        }

        // online softmax: row max / rescale factors (16-lane reductions)
        float alpha[8], mnew[8];
#pragma unroll
        for (int j = 0; j < 8; ++j) {
            float mx = fmaxf(s[0][j], s[1][j]);
#pragma unroll
            for (int off = 1; off < 16; off <<= 1)
                mx = fmaxf(mx, __shfl_xor(mx, off, 32));
            const float mn = fmaxf(mst[j], mx);
            mnew[j] = mn;
            alpha[j] = __expf(mst[j] - mn);
            mst[j] = mn;
        }

        // p = exp(s - mnew); accumulate row sums; stage p (bf16) in LDS
#pragma unroll
        for (int j = 0; j < 8; ++j) {
            float ps = 0.0f;
#pragma unroll
            for (int t = 0; t < 2; ++t) {
                const float p = __expf(s[t][j] - mnew[j]);
                ps += p;
                pst[(j + 8 * half) * 32 + t * 16 + ln] = (__bf16)p;
            }
#pragma unroll
            for (int off = 1; off < 16; off <<= 1)
                ps += __shfl_xor(ps, off, 32);
            lst[j] = lst[j] * alpha[j] + ps;
        }

        // rescale running output
#pragma unroll
        for (int t = 0; t < 4; ++t)
#pragma unroll
            for (int j = 0; j < 8; ++j) accO[t][j] *= alpha[j];

        // LDS writes from this wave must land before re-reading in A layout
        asm volatile("s_wait_dscnt 0" ::: "memory");

        const __bf16* prow = pst + ln * 32;
    const v16bf ap = load_frag16(prow + half * 16, prow + half * 16 + 8);

        // O += P * V  (V stored transposed [v][m] -> contiguous B loads)
#pragma unroll
        for (int t = 0; t < 4; ++t) {
            const __bf16* vrow = vh + (size_t)(t * 16 + ln) * NM + n0 + half * 16;
            accO[t] = wmma_bf16(ap, load_frag16(vrow, vrow + 8), accO[t]);
        }
    }

    // epilogue: O /= l, store as o_flat [b][m][h*64+v] (bf16)
    float inv[8];
#pragma unroll
    for (int j = 0; j < 8; ++j) inv[j] = __builtin_amdgcn_rcpf(lst[j]);
    __bf16* obase = ob + (size_t)b * NM * (NH * NV) + (size_t)h * NV;
#pragma unroll
    for (int t = 0; t < 4; ++t)
#pragma unroll
        for (int j = 0; j < 8; ++j)
            obase[(size_t)(m0 + j + 8 * half) * (NH * NV) + t * 16 + ln] =
                (__bf16)(accO[t][j] * inv[j]);
}

// ---------------------------------------------------------------------------
// Output projection: y[b] = o_flat[b] ([M x 1024]) @ W ([1024 x 1024]), fp32 out
// ---------------------------------------------------------------------------
__global__ __launch_bounds__(256) void oproj_kernel(
    const __bf16* __restrict__ ob,
    const __bf16* __restrict__ wot,
    float* __restrict__ y) {
    const int lane = threadIdx.x & 31;
    const int wave = threadIdx.x >> 5;
    const int ln   = lane & 15;
    const int half = lane >> 4;
    const int b  = blockIdx.z;
    const int m0 = blockIdx.y * 128 + wave * 16;
    const int d0 = blockIdx.x * 64;
    const int C  = NH * NV;  // 1024

    const __bf16* arow = ob + ((size_t)b * NM + m0 + ln) * C;
    v8f acc[4];
#pragma unroll
    for (int t = 0; t < 4; ++t) acc[t] = v8f_zero();

    for (int c0 = 0; c0 < C; c0 += 32) {
        v16bf a = load_frag16(arow + c0 + half * 16, arow + c0 + half * 16 + 8);
#pragma unroll
        for (int t = 0; t < 4; ++t) {
            const __bf16* brow = wot + (size_t)(d0 + t * 16 + ln) * C + c0 + half * 16;
            acc[t] = wmma_bf16(a, load_frag16(brow, brow + 8), acc[t]);
        }
    }
#pragma unroll
    for (int t = 0; t < 4; ++t)
#pragma unroll
        for (int j = 0; j < 8; ++j)
            y[((size_t)b * NM + m0 + j + 8 * half) * ND + d0 + t * 16 + ln] =
                acc[t][j];
}

// ---------------------------------------------------------------------------
extern "C" void kernel_launch(void* const* d_in, const int* in_sizes, int n_in,
                              void* d_out, int out_size, void* d_ws, size_t ws_size,
                              hipStream_t stream) {
    (void)in_sizes; (void)n_in; (void)out_size; (void)ws_size;
    const float* x  = (const float*)d_in[0];
    const float* Pq = (const float*)d_in[1];
    const float* Pk = (const float*)d_in[2];
    const float* Pv = (const float*)d_in[3];
    const float* Po = (const float*)d_in[4];
    float* y = (float*)d_out;

    __bf16* ws = (__bf16*)d_ws;
    size_t off = 0;
    __bf16* xb  = ws + off; off += (size_t)NB * NM * ND;   // x bf16
    __bf16* pqt = ws + off; off += (size_t)NH * NK * ND;   // Pq^T
    __bf16* pkt = ws + off; off += (size_t)NH * NK * ND;   // Pk^T
    __bf16* pvt = ws + off; off += (size_t)NH * NK * ND;   // Pv^T
    __bf16* wot = ws + off; off += (size_t)ND * NH * NV;   // Wo^T
    __bf16* qb  = ws + off; off += (size_t)NB * NH * NM * NK;  // Q
    __bf16* kb  = ws + off; off += (size_t)NB * NH * NM * NK;  // K
    __bf16* vtb = ws + off; off += (size_t)NB * NH * NV * NM;  // V^T
    __bf16* ob  = ws + off; off += (size_t)NB * NM * NH * NV;  // o_flat

    pack_x_kernel<<<dim3(2048), dim3(256), 0, stream>>>(x, xb, NB * NM * ND);
    pack_w_kernel<<<dim3(512), dim3(256), 0, stream>>>(Pq, pqt);
    pack_w_kernel<<<dim3(512), dim3(256), 0, stream>>>(Pk, pkt);
    pack_w_kernel<<<dim3(512), dim3(256), 0, stream>>>(Pv, pvt);
    pack_wo_kernel<<<dim3(512), dim3(256), 0, stream>>>(Po, wot);

    qkv_kernel<<<dim3(NM / 128, NB * NH), dim3(256), 0, stream>>>(
        xb, pqt, pkt, pvt, qb, kb, vtb);
    attn_kernel<<<dim3(NM / 128, NB * NH), dim3(256), 0, stream>>>(
        qb, kb, vtb, ob);
    oproj_kernel<<<dim3(ND / 64, NM / 128, NB), dim3(256), 0, stream>>>(
        ob, wot, y);
}
